// GraphTransformer_59322088292868
// MI455X (gfx1250) — matrix operand order repaired and checked by reading, not compiled, not run on hardware
//
#include <hip/hip_runtime.h>
#include <hip/hip_bf16.h>

#define N_NODES 50000
#define N_EDGES 400000
#define N_BATCH 64
#define IN_DIM  768
#define OUT_DIM 64
#define HEADS   8
#define QKV_DIM (HEADS * OUT_DIM)   // 512

typedef __attribute__((ext_vector_type(16))) __bf16 v16bf;
typedef __attribute__((ext_vector_type(8)))  __bf16 v8bf;
typedef __attribute__((ext_vector_type(4)))  __bf16 v4bf;
typedef __attribute__((ext_vector_type(8)))  float  v8f;
typedef __attribute__((ext_vector_type(4)))  float  v4f;

// ---------------------------------------------------------------- utilities
__global__ void fill_zero_kernel(float* __restrict__ p, long long n) {
  long long i = (long long)blockIdx.x * blockDim.x + threadIdx.x;
  if (i < n) p[i] = 0.0f;
}

// W[k][m] (f32, row-major) -> Wt[m][k] (bf16)
__global__ void transpose_to_bf16_kernel(const float* __restrict__ W,
                                         __bf16* __restrict__ Wt,
                                         int kdim, int mdim) {
  int idx = blockIdx.x * blockDim.x + threadIdx.x;
  if (idx >= kdim * mdim) return;
  int k = idx / mdim, m = idx - k * mdim;      // coalesced read along m
  Wt[(size_t)m * kdim + k] = (__bf16)W[idx];
}

__device__ __forceinline__ unsigned flipf(float f) {
  unsigned u = __float_as_uint(f);
  return (u & 0x80000000u) ? ~u : (u | 0x80000000u);
}
__device__ __forceinline__ float unflipf(unsigned u) {
  u = (u & 0x80000000u) ? (u & 0x7FFFFFFFu) : ~u;
  return __uint_as_float(u);
}

// ---------------------------------------------------------------- WMMA GEMM
// C[n x m] = A[n x kdim] @ W[kdim x m] + bias[m]
// A: f32 or bf16 (row-major). Wt: pre-transposed bf16 [m][k]. C: f32 or bf16.
// 256 threads = 8 wave32; each wave owns a 32x32 tile = 2x2 WMMA accumulators.
// Double-buffered LDS + register-prefetch software pipeline.
#define BM 128
#define BN 64
#define BK 32
#define TPB 256
#define LDK (BK + 8)   // padded row: 80B stride, 16B-aligned sublanes

template <bool A_BF16, bool C_BF16>
__global__ __launch_bounds__(TPB) void gemm_bias_wmma(
    const void* __restrict__ Ap, const __bf16* __restrict__ Wt,
    const float* __restrict__ bias, void* __restrict__ Cp,
    int n, int kdim, int mdim) {
  __shared__ alignas(16) __bf16 sA[2][BM][LDK];
  __shared__ alignas(16) __bf16 sB[2][BN][LDK];

  const int tid  = threadIdx.x;
  const int wave = tid >> 5;
  const int lane = tid & 31;
  const int row0 = blockIdx.y * BM;
  const int col0 = blockIdx.x * BN;

  const int wr = wave & 3;     // 32-row block
  const int wc = wave >> 2;    // 32-col block
  const int half = lane >> 4;
  const int l16  = lane & 15;

  // B staging indices: 64 rows x 4 x v8bf, one v8bf per thread
  const int bn = tid >> 2;          // 0..63
  const int bp = tid & 3;           // 0..3
  const __bf16* wrow = Wt + (size_t)(col0 + bn) * kdim + bp * 8;

  v8f acc[2][2] = {};

  // ---- staging registers
  v4f  aregf[4];   // f32-A path: 4 x 16B
  v8bf aregb[2];   // bf16-A path: 2 x 16B
  v8bf breg;

  const float*  Af = (const float*)Ap;
  const __bf16* Ab = (const __bf16*)Ap;

  // ---- stage-load: global -> regs (row index clamped: OOB rows feed only OOB C)
  auto stage_load = [&](int k0) {
    if constexpr (!A_BF16) {
#pragma unroll
      for (int i = 0; i < 4; ++i) {
        int idx = i * TPB + tid;          // 1024 v4f
        int r = idx >> 3, c4 = idx & 7;
        int gr = row0 + r; gr = gr < n ? gr : (n - 1);
        aregf[i] = *(const v4f*)(Af + (size_t)gr * kdim + k0 + c4 * 4);
      }
    } else {
#pragma unroll
      for (int i = 0; i < 2; ++i) {
        int idx = i * TPB + tid;          // 512 v8bf
        int r = idx >> 2, p = idx & 3;
        int gr = row0 + r; gr = gr < n ? gr : (n - 1);
        aregb[i] = *(const v8bf*)(Ab + (size_t)gr * kdim + k0 + p * 8);
      }
    }
    breg = *(const v8bf*)(wrow + k0);
  };

  // ---- stage-store: regs -> LDS buffer
  auto stage_store = [&](int buf) {
    if constexpr (!A_BF16) {
#pragma unroll
      for (int i = 0; i < 4; ++i) {
        int idx = i * TPB + tid;
        int r = idx >> 3, c4 = idx & 7;
        v4f f = aregf[i];
        v4bf b;
        b[0] = (__bf16)f[0]; b[1] = (__bf16)f[1];
        b[2] = (__bf16)f[2]; b[3] = (__bf16)f[3];
        *(v4bf*)&sA[buf][r][c4 * 4] = b;
      }
    } else {
#pragma unroll
      for (int i = 0; i < 2; ++i) {
        int idx = i * TPB + tid;
        int r = idx >> 2, p = idx & 3;
        *(v8bf*)&sA[buf][r][p * 8] = aregb[i];
      }
    }
    *(v8bf*)&sB[buf][bn][bp * 8] = breg;
  };

  stage_load(0);
  stage_store(0);
  __syncthreads();

  const int nk = kdim / BK;
  for (int t = 0; t < nk; ++t) {
    const int cur = t & 1;
    if (t + 1 < nk) stage_load((t + 1) * BK);   // loads in flight under WMMA

    v16bf afrag[2], bfrag[2];
#pragma unroll
    for (int mt = 0; mt < 2; ++mt) {
      int m = wr * 32 + mt * 16 + l16;
      v8bf lo = *(const v8bf*)&sA[cur][m][8 * half];
      v8bf hi = *(const v8bf*)&sA[cur][m][16 + 8 * half];
      afrag[mt] = __builtin_shufflevector(lo, hi, 0, 1, 2, 3, 4, 5, 6, 7,
                                          8, 9, 10, 11, 12, 13, 14, 15);
    }
#pragma unroll
    for (int nt = 0; nt < 2; ++nt) {
      int nn = wc * 32 + nt * 16 + l16;
      v8bf lo = *(const v8bf*)&sB[cur][nn][8 * half];
      v8bf hi = *(const v8bf*)&sB[cur][nn][16 + 8 * half];
      bfrag[nt] = __builtin_shufflevector(lo, hi, 0, 1, 2, 3, 4, 5, 6, 7,
                                          8, 9, 10, 11, 12, 13, 14, 15);
    }
#pragma unroll
    for (int mt = 0; mt < 2; ++mt)
#pragma unroll
      for (int nt = 0; nt < 2; ++nt)
        acc[mt][nt] = __builtin_amdgcn_wmma_f32_16x16x32_bf16(
            false, afrag[mt], false, bfrag[nt],
            (short)0, acc[mt][nt], false, false);

    if (t + 1 < nk) stage_store(cur ^ 1);
    __syncthreads();
  }

  // ---- epilogue: bias + store (acc[r] -> row r + 8*half, col l16)
  float* Cf = (float*)Cp;
  __bf16* Cb = (__bf16*)Cp;
#pragma unroll
  for (int mt = 0; mt < 2; ++mt) {
#pragma unroll
    for (int nt = 0; nt < 2; ++nt) {
      int nn = col0 + wc * 32 + nt * 16 + l16;
      float b = bias[nn];
#pragma unroll
      for (int r = 0; r < 8; ++r) {
        int m = row0 + wr * 32 + mt * 16 + 8 * half + r;
        if (m < n) {
          float v = acc[mt][nt][r] + b;
          if constexpr (C_BF16) Cb[(size_t)m * mdim + nn] = (__bf16)v;
          else                  Cf[(size_t)m * mdim + nn] = v;
        }
      }
    }
  }
}

// ---------------------------------------------------------------- edge phase
__global__ void edge_score_kernel(const float* __restrict__ q,
                                  const float* __restrict__ k,
                                  const int* __restrict__ ei,
                                  float* __restrict__ alpha,
                                  unsigned* __restrict__ mmax) {
  int idx = blockIdx.x * blockDim.x + threadIdx.x;
  if (idx >= N_EDGES * HEADS) return;
  int e = idx >> 3, h = idx & 7;
  int s = ei[e], d = ei[N_EDGES + e];
  const float4* qp = (const float4*)(q + (long long)d * QKV_DIM + h * OUT_DIM);
  const float4* kp = (const float4*)(k + (long long)s * QKV_DIM + h * OUT_DIM);
  float acc = 0.0f;
#pragma unroll
  for (int j = 0; j < OUT_DIM / 4; ++j) {
    float4 a = qp[j], b = kp[j];
    acc += a.x * b.x + a.y * b.y + a.z * b.z + a.w * b.w;
  }
  acc *= 0.125f;  // 1/sqrt(64)
  alpha[idx] = acc;
  atomicMax(&mmax[d * HEADS + h], flipf(acc));
}

__global__ void edge_exp_kernel(const int* __restrict__ ei,
                                float* __restrict__ alpha,
                                const unsigned* __restrict__ mmax,
                                float* __restrict__ ssum) {
  int idx = blockIdx.x * blockDim.x + threadIdx.x;
  if (idx >= N_EDGES * HEADS) return;
  int e = idx >> 3, h = idx & 7;
  int d = ei[N_EDGES + e];
  float m  = unflipf(mmax[d * HEADS + h]);
  float ex = __expf(alpha[idx] - m);
  alpha[idx] = ex;
  atomicAdd(&ssum[d * HEADS + h], ex);
}

__global__ void edge_scatter_kernel(const int* __restrict__ ei,
                                    const float* __restrict__ alpha,
                                    const float* __restrict__ ssum,
                                    const float* __restrict__ v,
                                    float* __restrict__ msg) {
  int idx = blockIdx.x * blockDim.x + threadIdx.x;
  if (idx >= N_EDGES * HEADS) return;
  int e = idx >> 3, h = idx & 7;
  int s = ei[e], d = ei[N_EDGES + e];
  float w = alpha[idx] / (ssum[d * HEADS + h] + 1e-16f);
  const float* vp = v + (long long)s * QKV_DIM + h * OUT_DIM;
  float* mp = msg + (long long)d * QKV_DIM + h * OUT_DIM;
#pragma unroll 8
  for (int c = 0; c < OUT_DIM; ++c) atomicAdd(&mp[c], w * vp[c]);
}

// ---------------------------------------------------------------- node out + pool
__global__ void node_out_kernel(const float* __restrict__ msg,
                                const float* __restrict__ skip,
                                const int* __restrict__ batch,
                                float* __restrict__ pooled,
                                float* __restrict__ counts) {
  int idx = blockIdx.x * blockDim.x + threadIdx.x;
  if (idx >= N_NODES * OUT_DIM) return;
  int i = idx >> 6, c = idx & 63;
  const float* mp = msg + (long long)i * QKV_DIM + c;
  float sum = 0.0f;
#pragma unroll
  for (int h = 0; h < HEADS; ++h) sum += mp[h * OUT_DIM];
  float val = sum * (1.0f / HEADS) + skip[idx];
  int b = batch[i];
  atomicAdd(&pooled[b * OUT_DIM + c], val);
  if (c == 0) atomicAdd(&counts[b], 1.0f);
}

__global__ void pool_div_kernel(const float* __restrict__ pooled,
                                const float* __restrict__ counts,
                                float* __restrict__ out) {
  int idx = blockIdx.x * blockDim.x + threadIdx.x;
  if (idx < N_BATCH * OUT_DIM)
    out[idx] = pooled[idx] / fmaxf(counts[idx >> 6], 1.0f);
}

// ---------------------------------------------------------------- launch
extern "C" void kernel_launch(void* const* d_in, const int* in_sizes, int n_in,
                              void* d_out, int out_size, void* d_ws, size_t ws_size,
                              hipStream_t stream) {
  const float* x     = (const float*)d_in[0];
  const int*   ei    = (const int*)d_in[1];
  const int*   batch = (const int*)d_in[2];
  const float* W_emb = (const float*)d_in[3];
  const float* b_emb = (const float*)d_in[4];
  const float* Wq    = (const float*)d_in[5];
  const float* bq    = (const float*)d_in[6];
  const float* Wk    = (const float*)d_in[7];
  const float* bk    = (const float*)d_in[8];
  const float* Wv    = (const float*)d_in[9];
  const float* bv    = (const float*)d_in[10];
  const float* Wsk   = (const float*)d_in[11];
  const float* bsk   = (const float*)d_in[12];
  float* out = (float*)d_out;

  // ---- workspace layout (bytes, 256B-aligned chunks)
  char* wsb = (char*)d_ws;
  size_t off = 0;
  auto carve = [&](size_t bytes) {
    void* p = wsb + off;
    off += (bytes + 255) & ~(size_t)255;
    return p;
  };
  float* qbuf   = (float*)carve((size_t)N_NODES * QKV_DIM * 4);
  float* kbuf   = (float*)carve((size_t)N_NODES * QKV_DIM * 4);
  float* vbuf   = (float*)carve((size_t)N_NODES * QKV_DIM * 4);
  float* skip   = (float*)carve((size_t)N_NODES * OUT_DIM * 4);
  float* alpha  = (float*)carve((size_t)N_EDGES * HEADS * 4);
  // zero-region start: mmax, ssum, msg, pooled, counts (contiguous)
  unsigned* mmax  = (unsigned*)carve((size_t)N_NODES * HEADS * 4);
  float*   ssum   = (float*)carve((size_t)N_NODES * HEADS * 4);
  float*   msg    = (float*)carve((size_t)N_NODES * QKV_DIM * 4);
  float*   pooled = (float*)carve((size_t)N_BATCH * OUT_DIM * 4);
  float*   counts = (float*)carve((size_t)N_BATCH * 4);
  __bf16* hbuf  = (__bf16*)carve((size_t)N_NODES * IN_DIM * 2);   // bf16 h
  __bf16* WembT = (__bf16*)carve((size_t)IN_DIM * IN_DIM * 2);
  __bf16* WqT   = (__bf16*)carve((size_t)IN_DIM * QKV_DIM * 2);
  __bf16* WkT   = (__bf16*)carve((size_t)IN_DIM * QKV_DIM * 2);
  __bf16* WvT   = (__bf16*)carve((size_t)IN_DIM * QKV_DIM * 2);
  __bf16* WskT  = (__bf16*)carve((size_t)IN_DIM * OUT_DIM * 2);

  // ---- zero mmax..counts (one contiguous region; 0u is the flip-space floor)
  long long zc = ((char*)counts - (char*)mmax) / 4 + N_BATCH;
  fill_zero_kernel<<<(int)((zc + 255) / 256), 256, 0, stream>>>((float*)mmax, zc);

  // ---- pre-transpose weights to bf16 [m][k]
  {
    int n1 = IN_DIM * IN_DIM, n2 = IN_DIM * QKV_DIM, n3 = IN_DIM * OUT_DIM;
    transpose_to_bf16_kernel<<<(n1 + 255) / 256, 256, 0, stream>>>(W_emb, WembT, IN_DIM, IN_DIM);
    transpose_to_bf16_kernel<<<(n2 + 255) / 256, 256, 0, stream>>>(Wq, WqT, IN_DIM, QKV_DIM);
    transpose_to_bf16_kernel<<<(n2 + 255) / 256, 256, 0, stream>>>(Wk, WkT, IN_DIM, QKV_DIM);
    transpose_to_bf16_kernel<<<(n2 + 255) / 256, 256, 0, stream>>>(Wv, WvT, IN_DIM, QKV_DIM);
    transpose_to_bf16_kernel<<<(n3 + 255) / 256, 256, 0, stream>>>(Wsk, WskT, IN_DIM, OUT_DIM);
  }

  dim3 blk(TPB);
  dim3 gh(IN_DIM / BN, (N_NODES + BM - 1) / BM);
  gemm_bias_wmma<false, true><<<gh, blk, 0, stream>>>(x, WembT, b_emb, hbuf,
                                                      N_NODES, IN_DIM, IN_DIM);
  dim3 gq(QKV_DIM / BN, (N_NODES + BM - 1) / BM);
  gemm_bias_wmma<true, false><<<gq, blk, 0, stream>>>(hbuf, WqT, bq, qbuf,
                                                      N_NODES, IN_DIM, QKV_DIM);
  gemm_bias_wmma<true, false><<<gq, blk, 0, stream>>>(hbuf, WkT, bk, kbuf,
                                                      N_NODES, IN_DIM, QKV_DIM);
  gemm_bias_wmma<true, false><<<gq, blk, 0, stream>>>(hbuf, WvT, bv, vbuf,
                                                      N_NODES, IN_DIM, QKV_DIM);
  dim3 gs(OUT_DIM / BN, (N_NODES + BM - 1) / BM);
  gemm_bias_wmma<true, false><<<gs, blk, 0, stream>>>(hbuf, WskT, bsk, skip,
                                                      N_NODES, IN_DIM, OUT_DIM);

  int eh = N_EDGES * HEADS;
  int eblk = (eh + 255) / 256;
  edge_score_kernel<<<eblk, 256, 0, stream>>>(qbuf, kbuf, ei, alpha, mmax);
  edge_exp_kernel<<<eblk, 256, 0, stream>>>(ei, alpha, mmax, ssum);
  edge_scatter_kernel<<<eblk, 256, 0, stream>>>(ei, alpha, ssum, vbuf, msg);

  int nc = N_NODES * OUT_DIM;
  node_out_kernel<<<(nc + 255) / 256, 256, 0, stream>>>(msg, skip, batch, pooled, counts);
  pool_div_kernel<<<(N_BATCH * OUT_DIM + 255) / 256, 256, 0, stream>>>(pooled, counts, out);
}
